// MultiHeadAttention_81054622810713
// MI455X (gfx1250) — matrix operand order repaired and checked by reading, not compiled, not run on hardware
//
#include <hip/hip_runtime.h>
#include <hip/hip_bf16.h>
#include <math.h>

// ---------------------------------------------------------------------------
// MI455X (gfx1250) multi-head attention forward.
//  - bf16 WMMA (v_wmma_f32_16x16x32_bf16) for all matmuls, f32 accumulate
//  - all GEMM/attention operands pre-converted to bf16 in scratch (half HBM
//    traffic, enables async LDS staging)
//  - GLOBAL_LOAD_ASYNC_TO_LDS_B128 + s_wait_asynccnt for non-transposed tiles
//  - flash-attention single pass (scores never hit HBM)
// ---------------------------------------------------------------------------

typedef __attribute__((ext_vector_type(16))) __bf16 v16bf;
typedef __attribute__((ext_vector_type(8)))  float  v8f;

union FragBF { v16bf v; unsigned int u[8]; };
union Pack8  { uint4 q; unsigned short s[8]; };

__device__ __forceinline__ unsigned short f2bf(float f) {
    union { float f; unsigned int u; } x; x.f = f;
    unsigned int r = x.u + 0x7FFFu + ((x.u >> 16) & 1u);  // round-to-nearest-even
    return (unsigned short)(r >> 16);
}

// Async 16-byte global -> LDS copy (ASYNCcnt-tracked, gfx1250 TDM-lite path).
__device__ __forceinline__ void async_copy_b128(const void* gptr, void* lds_generic) {
    unsigned lds_off = (unsigned)(size_t)lds_generic;        // low 32 bits = LDS offset
    unsigned long long ga = (unsigned long long)gptr;
    asm volatile("global_load_async_to_lds_b128 %0, %1, off"
                 :: "v"(lds_off), "v"(ga) : "memory");
}
__device__ __forceinline__ void wait_async0() {
    asm volatile("s_wait_asynccnt 0" ::: "memory");
}

// ---------------------------------------------------------------------------
// fp32 -> bf16 elementwise convert (4 elems/thread, vectorized)
// ---------------------------------------------------------------------------
__global__ __launch_bounds__(256)
void mha_f32_to_bf16_kernel(const float* __restrict__ src,
                            unsigned short* __restrict__ dst, int n)
{
    int i = (blockIdx.x * 256 + threadIdx.x) * 4;
    if (i < n) {
        float4 f = *(const float4*)(src + i);
        union { unsigned short s[4]; uint2 u; } o;
        o.s[0] = f2bf(f.x); o.s[1] = f2bf(f.y);
        o.s[2] = f2bf(f.z); o.s[3] = f2bf(f.w);
        *(uint2*)(dst + i) = o.u;
    }
}

// ---------------------------------------------------------------------------
// GEMM: C[M,N] = A[M,K] @ B[K,N] + bias[N]   (A,B bf16; C bf16 or fp32)
// Block: 256 threads = 8 waves (2 M-waves x 4 N-waves). Block tile 32x128.
// Each wave computes two 16x16 WMMA tiles sharing one A fragment. K-step 32.
// A tile staged with async-to-LDS copies; B tile transposed via LDS stores.
// ---------------------------------------------------------------------------
#define GK 32
__global__ __launch_bounds__(256)
void mha_gemm_bias_kernel(const unsigned short* __restrict__ A,   // [M,K] bf16
                          const unsigned short* __restrict__ Bw,  // [K,N] bf16
                          const float* __restrict__ bias,
                          void* __restrict__ Cout,
                          int M, int N, int K, int c_bf16)
{
    __shared__ unsigned short As[32][GK];    // [m][k]  2 KB
    __shared__ unsigned short Bs[128][GK];   // [n][k]  8 KB (transposed)

    const int tid    = threadIdx.x;
    const int wave   = tid >> 5;
    const int lane   = tid & 31;
    const int lane16 = lane & 15;
    const int half   = lane >> 4;
    const int wm     = wave >> 2;            // 0..1  (M sub-block)
    const int wn     = wave & 3;             // 0..3  (N sub-block, 32 wide)

    const int m0    = blockIdx.x * 32;
    const int n0blk = blockIdx.y * 128;

    v8f acc0 = {}, acc1 = {};

    for (int k0 = 0; k0 < K; k0 += GK) {
        // ---- A tile (32x32 bf16 = 2 KB): 128 async b128 copies -------------
        if (tid < 128) {
            int row = tid >> 2, seg = (tid & 3) * 8;
            async_copy_b128(A + (size_t)(m0 + row) * K + k0 + seg, &As[row][seg]);
        }
        // ---- B tile (32x128 bf16): load b128 along n, transpose into LDS ---
        for (int i = tid; i < 512; i += 256) {
            int kk = i >> 4, ng = (i & 15) * 8;
            Pack8 d;
            d.q = *(const uint4*)(Bw + (size_t)(k0 + kk) * N + n0blk + ng);
            #pragma unroll
            for (int j = 0; j < 8; ++j) Bs[ng + j][kk] = d.s[j];
        }
        if (k0 + GK < K)   // prefetch next weight tile into L2
            __builtin_prefetch(Bw + (size_t)(k0 + GK) * N + n0blk, 0, 1);

        wait_async0();
        __syncthreads();

        // ---- fragments + 2 WMMAs (A fragment reused) -----------------------
        FragBF a, b0, b1;
        #pragma unroll
        for (int v = 0; v < 8; ++v) {
            int ka = 16 * half + 2 * v;
            a.u[v]  = *(const unsigned int*)&As[wm * 16 + lane16][ka];
            int kb = 16 * half + 2 * v;
            b0.u[v] = *(const unsigned int*)&Bs[wn * 32 + lane16][kb];
            b1.u[v] = *(const unsigned int*)&Bs[wn * 32 + 16 + lane16][kb];
        }
        acc0 = __builtin_amdgcn_wmma_f32_16x16x32_bf16(false, a.v, false, b0.v, (short)0, acc0, false, false);
        acc1 = __builtin_amdgcn_wmma_f32_16x16x32_bf16(false, a.v, false, b1.v, (short)0, acc1, false, false);
        __syncthreads();
    }

    const int nA = n0blk + wn * 32 + lane16;
    const int nB = nA + 16;
    const float biasA = bias[nA];
    const float biasB = bias[nB];
    #pragma unroll
    for (int v = 0; v < 8; ++v) {
        int m = m0 + wm * 16 + v + 8 * half;        // C layout: VGPR v -> row
        float va = acc0[v] + biasA;
        float vb = acc1[v] + biasB;
        if (c_bf16) {
            ((unsigned short*)Cout)[(size_t)m * N + nA] = f2bf(va);
            ((unsigned short*)Cout)[(size_t)m * N + nB] = f2bf(vb);
        } else {
            ((float*)Cout)[(size_t)m * N + nA] = va;
            ((float*)Cout)[(size_t)m * N + nB] = vb;
        }
    }
}

// ---------------------------------------------------------------------------
// Flash attention (bf16 in / bf16 out), online softmax.
// Grid: (Lq/64, NUM_HEADS, B). Block: 128 threads = 4 waves; wave = 16 q-rows.
// Key tile = 32; d_k = 64. Q and K tiles async-staged; V transposed via LDS.
// ---------------------------------------------------------------------------
#define FKT 32
__global__ __launch_bounds__(128)
void mha_flash_attn_kernel(const unsigned short* __restrict__ Qp,
                           const unsigned short* __restrict__ Kp,
                           const unsigned short* __restrict__ Vp,
                           const unsigned char* __restrict__ mask,  // [B, Lk] bool
                           unsigned short* __restrict__ Out,        // [B, Lq, D] bf16
                           int Lq, int Lk, float inv_scale)
{
    __shared__ unsigned short Qs[64][64];        // [q][d]   8 KB
    __shared__ unsigned short Ks[FKT][64];       // [k][d]   4 KB
    __shared__ unsigned short Vt[64][FKT];       // [d][k]   4 KB (transposed)
    __shared__ unsigned short Ps[4][16][FKT];    // per-wave P staging, 4 KB

    const int tid    = threadIdx.x;
    const int wave   = tid >> 5;
    const int lane   = tid & 31;
    const int lane16 = lane & 15;
    const int half   = lane >> 4;

    const int qblk = blockIdx.x * 64;
    const int h    = blockIdx.y;
    const int b    = blockIdx.z;
    const int D    = 1024;

    const size_t baseQ = ((size_t)b * Lq + qblk) * D + (size_t)h * 64;
    const size_t baseK = (size_t)b * Lk * D + (size_t)h * 64;

    // ---- stage Q block (64 x 64 bf16 = 8 KB): 512 async b128 copies --------
    for (int i = tid; i < 512; i += 128) {
        int q = i >> 3, seg = (i & 7) * 8;
        async_copy_b128(Qp + baseQ + (size_t)q * D + seg, &Qs[q][seg]);
    }
    wait_async0();
    __syncthreads();

    // Q A-fragments for the two K-steps (d 0..31, 32..63); loop-invariant
    FragBF aq0, aq1;
    #pragma unroll
    for (int v = 0; v < 8; ++v) {
        int ka = 16 * half + 2 * v;
        aq0.u[v] = *(const unsigned int*)&Qs[wave * 16 + lane16][ka];
        aq1.u[v] = *(const unsigned int*)&Qs[wave * 16 + lane16][32 + ka];
    }

    const float NEG_INF = -__builtin_inff();
    float m_run[8], l_run[8];
    #pragma unroll
    for (int v = 0; v < 8; ++v) { m_run[v] = NEG_INF; l_run[v] = 0.0f; }
    v8f o[4] = {v8f{}, v8f{}, v8f{}, v8f{}};

    for (int kt = 0; kt < Lk; kt += FKT) {
        // ---- K tile async (32 x 64 bf16): 256 b128 copies ------------------
        for (int i = tid; i < 256; i += 128) {
            int kk = i >> 3, seg = (i & 7) * 8;
            async_copy_b128(Kp + baseK + (size_t)(kt + kk) * D + seg, &Ks[kk][seg]);
        }
        // ---- V tile transposed: b128 along d, scatter b16 into Vt ----------
        for (int i = tid; i < 256; i += 128) {
            int kk = i >> 3, seg = (i & 7) * 8;
            Pack8 d;
            d.q = *(const uint4*)(Vp + baseK + (size_t)(kt + kk) * D + seg);
            #pragma unroll
            for (int j = 0; j < 8; ++j) Vt[seg + j][kk] = d.s[j];
        }
        wait_async0();
        __syncthreads();

        // ---- S = Q @ K^T for 2 key sub-tiles of 16 -------------------------
        v8f s0 = {}, s1 = {};
        FragBF bkf;
        #pragma unroll
        for (int v = 0; v < 8; ++v) bkf.u[v] = *(const unsigned int*)&Ks[lane16][16 * half + 2 * v];
        s0 = __builtin_amdgcn_wmma_f32_16x16x32_bf16(false, aq0.v, false, bkf.v, (short)0, s0, false, false);
        #pragma unroll
        for (int v = 0; v < 8; ++v) bkf.u[v] = *(const unsigned int*)&Ks[lane16][32 + 16 * half + 2 * v];
        s0 = __builtin_amdgcn_wmma_f32_16x16x32_bf16(false, aq1.v, false, bkf.v, (short)0, s0, false, false);
        #pragma unroll
        for (int v = 0; v < 8; ++v) bkf.u[v] = *(const unsigned int*)&Ks[16 + lane16][16 * half + 2 * v];
        s1 = __builtin_amdgcn_wmma_f32_16x16x32_bf16(false, aq0.v, false, bkf.v, (short)0, s1, false, false);
        #pragma unroll
        for (int v = 0; v < 8; ++v) bkf.u[v] = *(const unsigned int*)&Ks[16 + lane16][32 + 16 * half + 2 * v];
        s1 = __builtin_amdgcn_wmma_f32_16x16x32_bf16(false, aq1.v, false, bkf.v, (short)0, s1, false, false);

        // ---- scale + mask (mask==True => -inf, matching reference) ---------
        const bool mk0 = mask[(size_t)b * Lk + kt + lane16] != 0;
        const bool mk1 = mask[(size_t)b * Lk + kt + 16 + lane16] != 0;
        #pragma unroll
        for (int v = 0; v < 8; ++v) {
            s0[v] = mk0 ? NEG_INF : s0[v] * inv_scale;
            s1[v] = mk1 ? NEG_INF : s1[v] * inv_scale;
        }

        // ---- online softmax (row = (v, half); keys vary across 16 lanes) ---
        float alpha[8], p0[8], p1[8];
        #pragma unroll
        for (int v = 0; v < 8; ++v) {
            float rm = fmaxf(s0[v], s1[v]);
            #pragma unroll
            for (int off = 1; off < 16; off <<= 1)
                rm = fmaxf(rm, __shfl_xor(rm, off, 32));
            float mn = fmaxf(m_run[v], rm);
            alpha[v] = (m_run[v] == mn) ? 1.0f : __expf(m_run[v] - mn);
            bool dead = (mn == NEG_INF);           // whole row masked so far
            p0[v] = dead ? 0.0f : __expf(s0[v] - mn);
            p1[v] = dead ? 0.0f : __expf(s1[v] - mn);
            float rs = p0[v] + p1[v];
            #pragma unroll
            for (int off = 1; off < 16; off <<= 1)
                rs += __shfl_xor(rs, off, 32);
            l_run[v] = l_run[v] * alpha[v] + rs;
            m_run[v] = mn;
        }

        #pragma unroll
        for (int dt = 0; dt < 4; ++dt)
            #pragma unroll
            for (int v = 0; v < 8; ++v)
                o[dt][v] *= alpha[v];

        // ---- P: C-layout -> per-wave LDS -> A-layout -----------------------
        #pragma unroll
        for (int v = 0; v < 8; ++v) {
            Ps[wave][v + 8 * half][lane16]      = f2bf(p0[v]);
            Ps[wave][v + 8 * half][16 + lane16] = f2bf(p1[v]);
        }
        asm volatile("s_wait_dscnt 0" ::: "memory");   // wave-local LDS ordering

        FragBF ap;
        #pragma unroll
        for (int v = 0; v < 8; ++v) {
            int ka = 16 * half + 2 * v;
            ap.u[v] = *(const unsigned int*)&Ps[wave][lane16][ka];
        }

        // ---- O += P @ V  (4 sub-tiles of 16 V-columns) ---------------------
        #pragma unroll
        for (int dt = 0; dt < 4; ++dt) {
            FragBF bv;
            #pragma unroll
            for (int v = 0; v < 8; ++v)
                bv.u[v] = *(const unsigned int*)&Vt[dt * 16 + lane16][16 * half + 2 * v];
            o[dt] = __builtin_amdgcn_wmma_f32_16x16x32_bf16(
                        false, ap.v, false, bv.v, (short)0, o[dt], false, false);
        }
        __syncthreads();   // protect Ks/Vt reuse
    }

    // ---- epilogue: normalize, write bf16 [B, Lq, D] ------------------------
    #pragma unroll
    for (int dt = 0; dt < 4; ++dt) {
        #pragma unroll
        for (int v = 0; v < 8; ++v) {
            float l = l_run[v];
            float val = (l > 0.0f) ? (o[dt][v] / l) : 0.0f;   // fully-masked -> 0
            int q = qblk + wave * 16 + v + 8 * half;
            Out[((size_t)b * Lq + q) * D + (size_t)h * 64 + dt * 16 + lane16] = f2bf(val);
        }
    }
}

// ---------------------------------------------------------------------------
// Launch: 7 bf16 converts -> 3 projection GEMMs -> flash attention -> out GEMM
// Scratch (bf16): qb|kb|vb|Qp|Kp|Vp|Ao (8 MB each) + Wqb|Wkb|Wvb|Wob (2 MB
// each) = 64 MB total.
// ---------------------------------------------------------------------------
extern "C" void kernel_launch(void* const* d_in, const int* in_sizes, int n_in,
                              void* d_out, int out_size, void* d_ws, size_t ws_size,
                              hipStream_t stream) {
    (void)in_sizes; (void)n_in; (void)out_size; (void)ws_size;

    const float* query = (const float*)d_in[0];
    const float* key   = (const float*)d_in[1];
    const float* value = (const float*)d_in[2];
    const unsigned char* mask = (const unsigned char*)d_in[3];  // jax bool_ = 1 byte
    const float* Wq = (const float*)d_in[4];
    const float* bq = (const float*)d_in[5];
    const float* Wk = (const float*)d_in[6];
    const float* bk = (const float*)d_in[7];
    const float* Wv = (const float*)d_in[8];
    const float* bv = (const float*)d_in[9];
    const float* Wo = (const float*)d_in[10];
    const float* bo = (const float*)d_in[11];
    float* out = (float*)d_out;

    const int B = 2, Lq = 2048, Lk = 2048, D = 1024;
    const int M = B * Lq;                              // 4096
    const size_t actE = (size_t)M * D;                 // 4M elems
    const size_t wE   = (size_t)D * D;                 // 1M elems

    unsigned short* p   = (unsigned short*)d_ws;
    unsigned short* qb  = p;  p += actE;
    unsigned short* kb  = p;  p += actE;
    unsigned short* vb  = p;  p += actE;
    unsigned short* Qp  = p;  p += actE;
    unsigned short* Kp  = p;  p += actE;
    unsigned short* Vp  = p;  p += actE;
    unsigned short* Ao  = p;  p += actE;
    unsigned short* Wqb = p;  p += wE;
    unsigned short* Wkb = p;  p += wE;
    unsigned short* Wvb = p;  p += wE;
    unsigned short* Wob = p;  p += wE;

    dim3 cB(256, 1, 1);
    mha_f32_to_bf16_kernel<<<dim3((int)(actE / 4 / 256)), cB, 0, stream>>>(query, qb, (int)actE);
    mha_f32_to_bf16_kernel<<<dim3((int)(actE / 4 / 256)), cB, 0, stream>>>(key,   kb, (int)actE);
    mha_f32_to_bf16_kernel<<<dim3((int)(actE / 4 / 256)), cB, 0, stream>>>(value, vb, (int)actE);
    mha_f32_to_bf16_kernel<<<dim3((int)(wE / 4 / 256)),   cB, 0, stream>>>(Wq, Wqb, (int)wE);
    mha_f32_to_bf16_kernel<<<dim3((int)(wE / 4 / 256)),   cB, 0, stream>>>(Wk, Wkb, (int)wE);
    mha_f32_to_bf16_kernel<<<dim3((int)(wE / 4 / 256)),   cB, 0, stream>>>(Wv, Wvb, (int)wE);
    mha_f32_to_bf16_kernel<<<dim3((int)(wE / 4 / 256)),   cB, 0, stream>>>(Wo, Wob, (int)wE);

    dim3 gGrid(M / 32, D / 128, 1);                    // (128, 8)
    dim3 gBlk(256, 1, 1);
    mha_gemm_bias_kernel<<<gGrid, gBlk, 0, stream>>>(qb, Wqb, bq, Qp, M, D, D, 1);
    mha_gemm_bias_kernel<<<gGrid, gBlk, 0, stream>>>(kb, Wkb, bk, Kp, M, D, D, 1);
    mha_gemm_bias_kernel<<<gGrid, gBlk, 0, stream>>>(vb, Wvb, bv, Vp, M, D, D, 1);

    dim3 fGrid(Lq / 64, 16, B);                        // (32, 16, 2)
    dim3 fBlk(128, 1, 1);
    mha_flash_attn_kernel<<<fGrid, fBlk, 0, stream>>>(Qp, Kp, Vp, mask, Ao,
                                                      Lq, Lk, 0.125f /* 1/sqrt(64) */);

    mha_gemm_bias_kernel<<<gGrid, gBlk, 0, stream>>>(Ao, Wob, bo, out, M, D, D, 0);
}
